// VariationalGCNEncoder_50148038148378
// MI455X (gfx1250) — matrix operand order repaired and checked by reading, not compiled, not run on hardware
//
#include <hip/hip_runtime.h>
#include <hip/hip_bf16.h>

// ---------------------------------------------------------------------------
// Variational GCN encoder for MI455X (gfx1250, wave32, WMMA).
//   mu, logstd = GCN(x) with 4 GCN convs (last two share input, fused N=256).
// GEMMs run on v_wmma_f32_16x16x32_bf16 (bf16 A/B, f32 accum) with a 32x64
// wave tile (8 WMMAs per k-step); segment-sum aggregation uses
// global_atomic_add_f32 (the 51MB accumulator fits in the 192MB L2).
// ---------------------------------------------------------------------------

typedef __attribute__((ext_vector_type(16))) __bf16 v16bf;
typedef __attribute__((ext_vector_type(8)))  float  v8f;

union Frag {
    v16bf v;
    uint4 q[2];
};

namespace {
constexpr int NN  = 50000;   // nodes
constexpr int EE  = 800000;  // edges
constexpr int INC = 512;     // in channels
constexpr int CH  = 256;     // hidden channels (also fused mu|ls width)
}

// ---------------------- normalization (D^-1/2) ------------------------------
__global__ void deg_init_kernel(float* __restrict__ deg, int n) {
    int i = blockIdx.x * blockDim.x + threadIdx.x;
    if (i < n) deg[i] = 1.0f;   // self loop contributes 1
}

__global__ void deg_accum_kernel(const int* __restrict__ dst, float* __restrict__ deg, int e) {
    int i = blockIdx.x * blockDim.x + threadIdx.x;
    if (i < e) atomicAdd(&deg[dst[i]], 1.0f);
}

__global__ void deg_rsqrt_kernel(const float* __restrict__ deg, float* __restrict__ dinv, int n) {
    int i = blockIdx.x * blockDim.x + threadIdx.x;
    if (i < n) dinv[i] = rsqrtf(deg[i]);   // deg >= 1 always
}

// ---------------------- bf16 conversion / weight packing --------------------
__global__ void cvt_bf16_kernel(const float* __restrict__ in, __bf16* __restrict__ out, int n) {
    int i = blockIdx.x * blockDim.x + threadIdx.x;
    if (i < n) out[i] = (__bf16)in[i];
}

// Pack row-major W[K][N] (f32) into fragment-major bf16:
//   Wp[((k/32)*N + n)*32 + (k%32)]  -> each lane's 16 K-values are contiguous.
__global__ void pack_weight_kernel(const float* __restrict__ W, __bf16* __restrict__ Wp,
                                   int K, int N) {
    int i = blockIdx.x * blockDim.x + threadIdx.x;
    if (i >= K * N) return;
    int k = i / N, n = i % N;
    Wp[(((size_t)(k >> 5) * N + n) << 5) + (k & 31)] = (__bf16)W[i];
}

// Pack [Wmu | Wls] (each K x 128) into one fragment-major K x 256 buffer.
__global__ void pack_weight_dual_kernel(const float* __restrict__ Wa, const float* __restrict__ Wb,
                                        __bf16* __restrict__ Wp, int K) {
    int i = blockIdx.x * blockDim.x + threadIdx.x;
    if (i >= K * 256) return;
    int k = i / 256, n = i % 256;
    float v = (n < 128) ? Wa[k * 128 + n] : Wb[k * 128 + (n - 128)];
    Wp[(((size_t)(k >> 5) * 256 + n) << 5) + (k & 31)] = (__bf16)v;
}

// ---------------------- WMMA GEMM: C[MxN] = A[MxK] @ B[KxN] -----------------
// A: row-major bf16. B: fragment-major packed bf16. C: row-major f32.
// One wave computes a 32(M) x 64(N) tile: 2 A fragments, 4 shared B
// fragments, 8 f32 accumulators -> 8 WMMAs per 192B of L2 traffic.
// M % 16 == 0, N % 64 == 0, K % 32 == 0.
__global__ __launch_bounds__(256)
void wmma_gemm_kernel(const __bf16* __restrict__ A, const __bf16* __restrict__ Bp,
                      float* __restrict__ C, int M, int N, int K) {
    const int lane = threadIdx.x & 31;
    const int wid  = blockIdx.x * 8 + (threadIdx.x >> 5);
    const int nStrips = N >> 6;          // 64 cols per wave
    const int mTiles  = (M + 31) >> 5;   // 32 rows per wave
    if (wid >= mTiles * nStrips) return;

    const int m0 = (wid / nStrips) << 5;
    const int n0 = (wid % nStrips) << 6;
    const int g  = lane >> 4;    // lane group: selects K half [16g,16g+16) (contiguous order)
    const int ln = lane & 15;

    // A: M = lane%16. First 16-row half is always valid (M % 16 == 0);
    // second half may fall off the end on the last tile -> clamp loads,
    // guard stores.
    const bool hasM1 = (m0 + 16) < M;
    const __bf16* arow0 = A + (size_t)(m0 + ln) * K;
    const __bf16* arow1 = hasM1 ? (A + (size_t)(m0 + 16 + ln) * K) : arow0;

    v8f acc0[4] = {};
    v8f acc1[4] = {};

    for (int kb = 0; kb < K; kb += 32) {
        // A 16x32 bf16 fragment: group g holds K in [16g,16g+16) contiguous
        // (matches packed B order) -> two contiguous 16B runs per row half.
        Frag a0, a1;
        a0.q[0] = *reinterpret_cast<const uint4*>(arow0 + kb + 16 * g);
        a0.q[1] = *reinterpret_cast<const uint4*>(arow0 + kb + 16 * g + 8);
        a1.q[0] = *reinterpret_cast<const uint4*>(arow1 + kb + 16 * g);
        a1.q[1] = *reinterpret_cast<const uint4*>(arow1 + kb + 16 * g + 8);
        if (kb + 32 < K) {
            __builtin_prefetch(arow0 + kb + 32, 0, 1);  // global_prefetch_b8
            __builtin_prefetch(arow1 + kb + 32, 0, 1);
        }

        #pragma unroll
        for (int j = 0; j < 4; ++j) {
            // B 32x16 fragment: lane ln = column, group g -> K base (g*16),
            // 16 contiguous bf16 in the packed buffer. Shared by both M halves.
            const __bf16* bb = Bp + (((size_t)(kb >> 5) * N + (n0 + j * 16 + ln)) << 5) + (g << 4);
            Frag b;
            b.q[0] = reinterpret_cast<const uint4*>(bb)[0];
            b.q[1] = reinterpret_cast<const uint4*>(bb)[1];
            acc0[j] = __builtin_amdgcn_wmma_f32_16x16x32_bf16(
                          false, a0.v, false, b.v, (short)0, acc0[j], false, false);
            acc1[j] = __builtin_amdgcn_wmma_f32_16x16x32_bf16(
                          false, a1.v, false, b.v, (short)0, acc1[j], false, false);
        }
    }

    // D layout: VGPR r -> M = m0 + r + 8*g ; N = n0 + j*16 + ln
    #pragma unroll
    for (int j = 0; j < 4; ++j) {
        #pragma unroll
        for (int r = 0; r < 8; ++r) {
            C[(size_t)(m0 + r + 8 * g) * N + (n0 + j * 16 + ln)] = acc0[j][r];
        }
    }
    if (hasM1) {
        #pragma unroll
        for (int j = 0; j < 4; ++j) {
            #pragma unroll
            for (int r = 0; r < 8; ++r) {
                C[(size_t)(m0 + 16 + r + 8 * g) * N + (n0 + j * 16 + ln)] = acc1[j][r];
            }
        }
    }
}

// ---------------------- edge aggregation (segment_sum) ----------------------
// One thread per (edge, 4 channels): agg[dst] += t[src] * dinv[src]*dinv[dst]
__global__ void aggregate_kernel(const float* __restrict__ T, const int* __restrict__ src,
                                 const int* __restrict__ dst, const float* __restrict__ dinv,
                                 float* __restrict__ agg, int e, int C) {
    int gid = blockIdx.x * blockDim.x + threadIdx.x;
    int perEdge = C >> 2;
    if (gid >= e * perEdge) return;
    int ed = gid / perEdge;
    int c  = (gid % perEdge) << 2;
    int s = src[ed], d = dst[ed];
    float w = dinv[s] * dinv[d];
    float4 v = *reinterpret_cast<const float4*>(T + (size_t)s * C + c);
    float* out = agg + (size_t)d * C + c;
    atomicAdd(out + 0, v.x * w);
    atomicAdd(out + 1, v.y * w);
    atomicAdd(out + 2, v.z * w);
    atomicAdd(out + 3, v.w * w);
}

// ---------------------- epilogues -------------------------------------------
// h = relu(agg + t*dinv^2 + b), stored bf16 for next WMMA GEMM.
__global__ void epilogue_relu_kernel(const float* __restrict__ agg, const float* __restrict__ T,
                                     const float* __restrict__ dinv, const float* __restrict__ bias,
                                     __bf16* __restrict__ out, int n, int C) {
    int i = blockIdx.x * blockDim.x + threadIdx.x;
    if (i >= n * C) return;
    int node = i / C, c = i % C;
    float di = dinv[node];
    float v = agg[i] + T[i] * di * di + bias[c];
    out[i] = (__bf16)fmaxf(v, 0.0f);
}

// Final layer: split fused 256 channels into mu (first 128) and logstd.
__global__ void epilogue_out_kernel(const float* __restrict__ agg, const float* __restrict__ T,
                                    const float* __restrict__ dinv, const float* __restrict__ bmu,
                                    const float* __restrict__ bls, float* __restrict__ out, int n) {
    int i = blockIdx.x * blockDim.x + threadIdx.x;
    if (i >= n * 256) return;
    int node = i / 256, c = i % 256;
    float di = dinv[node];
    float v = agg[i] + T[i] * di * di;
    if (c < 128)
        out[(size_t)node * 128 + c] = v + bmu[c];
    else
        out[(size_t)n * 128 + (size_t)node * 128 + (c - 128)] = v + bls[c - 128];
}

// ---------------------------------------------------------------------------
extern "C" void kernel_launch(void* const* d_in, const int* in_sizes, int n_in,
                              void* d_out, int out_size, void* d_ws, size_t ws_size,
                              hipStream_t stream) {
    const float* x   = (const float*)d_in[0];
    const float* W1  = (const float*)d_in[1];
    const float* b1  = (const float*)d_in[2];
    const float* W2  = (const float*)d_in[3];
    const float* b2  = (const float*)d_in[4];
    const float* Wmu = (const float*)d_in[5];
    const float* bmu = (const float*)d_in[6];
    const float* Wls = (const float*)d_in[7];
    const float* bls = (const float*)d_in[8];
    const int*   ei  = (const int*)d_in[9];
    const int* srcE = ei;        // edge_index[0]
    const int* dstE = ei + EE;   // edge_index[1]

    char* ws = (char*)d_ws;
    size_t off = 0;
    auto carve = [&](size_t bytes) -> void* {
        void* p = ws + off;
        off = (off + bytes + 255) & ~(size_t)255;
        return p;
    };
    __bf16* xbf  = (__bf16*)carve((size_t)NN * INC * 2);  // 51.2 MB
    __bf16* hbf  = (__bf16*)carve((size_t)NN * CH * 2);   // 25.6 MB (h1, then h2)
    float*  T    = (float*) carve((size_t)NN * CH * 4);   // 51.2 MB (h@W per layer)
    float*  AGG  = (float*) carve((size_t)NN * CH * 4);   // 51.2 MB (segment sum)
    __bf16* wp1  = (__bf16*)carve((size_t)INC * CH * 2);
    __bf16* wp2  = (__bf16*)carve((size_t)CH * CH * 2);
    __bf16* wp3  = (__bf16*)carve((size_t)CH * CH * 2);
    float*  deg  = (float*) carve((size_t)NN * 4);
    float*  dinv = (float*) carve((size_t)NN * 4);

    const int TB = 256;
    auto blk = [](int n, int t) { return (n + t - 1) / t; };

    // ---- gcn_norm: deg = 1 + indegree, dinv = rsqrt(deg) ----
    deg_init_kernel<<<blk(NN, TB), TB, 0, stream>>>(deg, NN);
    deg_accum_kernel<<<blk(EE, TB), TB, 0, stream>>>(dstE, deg, EE);
    deg_rsqrt_kernel<<<blk(NN, TB), TB, 0, stream>>>(deg, dinv, NN);

    // ---- precision conversion + WMMA-fragment weight packing ----
    cvt_bf16_kernel<<<blk(NN * INC, TB), TB, 0, stream>>>(x, xbf, NN * INC);
    pack_weight_kernel<<<blk(INC * CH, TB), TB, 0, stream>>>(W1, wp1, INC, CH);
    pack_weight_kernel<<<blk(CH * CH, TB), TB, 0, stream>>>(W2, wp2, CH, CH);
    pack_weight_dual_kernel<<<blk(CH * CH, TB), TB, 0, stream>>>(Wmu, Wls, wp3, CH);

    const int gemmWaves  = ((NN + 31) / 32) * (CH / 64);   // 32x64 wave tiles
    const int gemmBlocks = blk(gemmWaves, 8);
    const int aggThreads = EE * (CH / 4);                  // 51.2M
    const size_t aggBytes = (size_t)NN * CH * 4;

    // ---- layer 1: relu(GCNConv(x, W1, b1)) ----
    wmma_gemm_kernel<<<gemmBlocks, TB, 0, stream>>>(xbf, wp1, T, NN, CH, INC);
    hipMemsetAsync(AGG, 0, aggBytes, stream);
    aggregate_kernel<<<blk(aggThreads, TB), TB, 0, stream>>>(T, srcE, dstE, dinv, AGG, EE, CH);
    epilogue_relu_kernel<<<blk(NN * CH, TB), TB, 0, stream>>>(AGG, T, dinv, b1, hbf, NN, CH);

    // ---- layer 2: relu(GCNConv(h1, W2, b2)) ----
    wmma_gemm_kernel<<<gemmBlocks, TB, 0, stream>>>(hbf, wp2, T, NN, CH, CH);
    hipMemsetAsync(AGG, 0, aggBytes, stream);
    aggregate_kernel<<<blk(aggThreads, TB), TB, 0, stream>>>(T, srcE, dstE, dinv, AGG, EE, CH);
    epilogue_relu_kernel<<<blk(NN * CH, TB), TB, 0, stream>>>(AGG, T, dinv, b2, hbf, NN, CH);

    // ---- layer 3 (fused mu|logstd): GCNConv(h2, [Wmu|Wls], [bmu|bls]) ----
    wmma_gemm_kernel<<<gemmBlocks, TB, 0, stream>>>(hbf, wp3, T, NN, CH, CH);
    hipMemsetAsync(AGG, 0, aggBytes, stream);
    aggregate_kernel<<<blk(aggThreads, TB), TB, 0, stream>>>(T, srcE, dstE, dinv, AGG, EE, CH);
    epilogue_out_kernel<<<blk(NN * CH, TB), TB, 0, stream>>>(AGG, T, dinv, bmu, bls,
                                                             (float*)d_out, NN);
}
